// GNNActorBase_7181185319506
// MI455X (gfx1250) — matrix operand order repaired and checked by reading, not compiled, not run on hardware
//
#include <hip/hip_runtime.h>

typedef __attribute__((ext_vector_type(16))) __bf16       v16bf;
typedef __attribute__((ext_vector_type(8)))  float        v8f;
typedef __attribute__((ext_vector_type(4)))  unsigned int vu4;

union ABf { v16bf v; vu4 u[2]; };
struct Tile { ABf a; ABf b[4]; };

// ---------------- degree / CSR build ----------------

__global__ void k_zero_i32(int* p, int n) {
    int i = blockIdx.x * blockDim.x + threadIdx.x;
    if (i < n) p[i] = 0;
}

__global__ void k_count(const long long* ei, int* cnt, int E) {
    int e = blockIdx.x * blockDim.x + threadIdx.x;
    if (e < E) atomicAdd(&cnt[(int)ei[(size_t)E + e]], 1);
}

// dis = rsqrt(cnt + 1)   (self-loop included; deg >= 1 always)
__global__ void k_dis(const int* cnt, float* dis, int n) {
    int i = blockIdx.x * blockDim.x + threadIdx.x;
    if (i < n) dis[i] = rsqrtf((float)cnt[i] + 1.0f);
}

// per-256-block exclusive scan; block totals to bsums
__global__ __launch_bounds__(256)
void k_scan_block(const int* in, int* out, int* bsums, int n) {
    __shared__ int sh[256];
    int tid = threadIdx.x;
    int i = blockIdx.x * 256 + tid;
    int v = (i < n) ? in[i] : 0;
    sh[tid] = v;
    __syncthreads();
    for (int ofs = 1; ofs < 256; ofs <<= 1) {
        int t = (tid >= ofs) ? sh[tid - ofs] : 0;
        __syncthreads();
        sh[tid] += t;
        __syncthreads();
    }
    if (i < n) out[i] = sh[tid] - v;            // exclusive within block
    if (tid == 255) bsums[blockIdx.x] = sh[tid];
}

// single-block exclusive scan of the (<=256) block sums, in place
__global__ __launch_bounds__(256)
void k_scan_tops(int* b, int n) {
    __shared__ int sh[256];
    int tid = threadIdx.x;
    int v = (tid < n) ? b[tid] : 0;
    sh[tid] = v;
    __syncthreads();
    for (int ofs = 1; ofs < 256; ofs <<= 1) {
        int t = (tid >= ofs) ? sh[tid - ofs] : 0;
        __syncthreads();
        sh[tid] += t;
        __syncthreads();
    }
    if (tid < n) b[tid] = sh[tid] - v;
}

__global__ void k_scan_add(int* rowStart, const int* boffs, int n, int E) {
    int i = blockIdx.x * blockDim.x + threadIdx.x;
    if (i < n)       rowStart[i] += boffs[i >> 8];
    else if (i == n) rowStart[n]  = E;
}

__global__ void k_copy_i32(const int* a, int* b, int n) {
    int i = blockIdx.x * blockDim.x + threadIdx.x;
    if (i < n) b[i] = a[i];
}

__global__ void k_place(const long long* ei, int* cursor, int* edgeSrc, int E) {
    int e = blockIdx.x * blockDim.x + threadIdx.x;
    if (e < E) {
        int s = (int)ei[e];
        int d = (int)ei[(size_t)E + e];
        int pos = atomicAdd(&cursor[d], 1);
        edgeSrc[pos] = s;
    }
}

// ---------------- conversions / weight repack ----------------

__global__ void k_cvt_bf16(const float* x, __bf16* y, size_t n) {
    size_t i = (size_t)blockIdx.x * blockDim.x + threadIdx.x;
    if (i < n) y[i] = (__bf16)x[i];
}

// Repack row-major fp32 W[K,N] into bf16 WMMA-B lane chunks:
// idx = ((kt*(N/16)+nt)*32 + lane)*16 + m  holds
// W[kt*32 + (lane>>4)*16 + m][nt*16 + (lane&15)].
__global__ void k_pack_w(const float* W, __bf16* Bp, int K, int N) {
    int idx = blockIdx.x * blockDim.x + threadIdx.x;
    if (idx >= K * N) return;
    int m    = idx & 15;
    int lane = (idx >> 4) & 31;
    int t    = idx >> 9;
    int NT   = N >> 4;
    int nt   = t % NT;
    int kt   = t / NT;
    int n    = nt * 16 + (lane & 15);
    int k    = kt * 32 + (lane >> 4) * 16 + m;
    Bp[idx]  = (__bf16)W[(size_t)k * N + n];
}

// ---------------- WMMA bf16 GEMM (ping-pong pipelined k-loop) ----------------
// One wave computes a 16(M) x 64(N) tile via 4 accumulators.

__device__ __forceinline__
void load_tile(Tile& t, const __bf16* arow, const __bf16* bbase,
               size_t kStride, int kt) {
    const __bf16* ap = arow + kt * 32;
    t.a.u[0] = *(const vu4*)(ap);
    t.a.u[1] = *(const vu4*)(ap + 8);
    const __bf16* bp = bbase + (size_t)kt * kStride;
#pragma unroll
    for (int j = 0; j < 4; ++j) {
        t.b[j].u[0] = *(const vu4*)(bp + j * 512);
        t.b[j].u[1] = *(const vu4*)(bp + j * 512 + 8);
    }
}

__device__ __forceinline__
void mma_tile(v8f acc[4], const Tile& t) {
#pragma unroll
    for (int j = 0; j < 4; ++j)
        acc[j] = __builtin_amdgcn_wmma_f32_16x16x32_bf16(
            false, t.a.v, false, t.b[j].v, (short)0, acc[j], false, false);
}

__global__ __launch_bounds__(256)
void k_gemm(const __bf16* __restrict__ A, const __bf16* __restrict__ Bp,
            const float* __restrict__ bias, float* outF, __bf16* outB,
            int M, int N, int K, int relu)
{
    int wid    = blockIdx.x * (blockDim.x >> 5) + (threadIdx.x >> 5);
    int nquads = N >> 6;
    int total  = (M >> 4) * nquads;
    if (wid >= total) return;                      // whole wave exits: EXEC all-1s
    int mt = wid / nquads, nq = wid % nquads;
    int lane = threadIdx.x & 31;
    int half = lane >> 4, r = lane & 15;

    const __bf16* arow  = A + (size_t)(mt * 16 + r) * K + half * 16;
    int NT = N >> 4;
    const size_t kStride = (size_t)NT * 512;       // packed-B elems per ktile
    const __bf16* bbase  = Bp + (size_t)(nq * 4) * 512 + (size_t)lane * 16;
    int ktiles = K >> 5;                           // always even (4 or 8)

    v8f acc[4] = {};
    Tile t0, t1;
    load_tile(t0, arow, bbase, kStride, 0);
    for (int kt = 0; kt < ktiles; kt += 2) {
        load_tile(t1, arow, bbase, kStride, kt + 1);
        mma_tile(acc, t0);                         // overlaps with t1's loads
        if (kt + 2 < ktiles)
            load_tile(t0, arow, bbase, kStride, kt + 2);
        mma_tile(acc, t1);                         // overlaps with t0's loads
    }

#pragma unroll
    for (int j = 0; j < 4; ++j) {
        int col = (nq * 4 + j) * 16 + r;
        float bv = bias ? bias[col] : 0.0f;
        int rowBase = mt * 16 + half * 8;
#pragma unroll
        for (int t = 0; t < 8; ++t) {
            float vv = acc[j][t] + bv;
            if (relu) vv = fmaxf(vv, 0.0f);
            size_t off = (size_t)(rowBase + t) * N + col;
            if (outF) outF[off] = vv;
            if (outB) outB[off] = (__bf16)vv;
        }
    }
}

// ---------------- fused CSR gather + bias + relu + residual -> bf16 ----------
// One wave per destination node: streams neighbor h-rows (512B coalesced),
// accumulates in registers; no atomics. Edge loop unrolled x2 so two row
// fetches are in flight per iteration.

__global__ __launch_bounds__(256)
void k_gather(const int* __restrict__ rowStart, const int* __restrict__ edgeSrc,
              const float* __restrict__ dis, const float* __restrict__ h,
              const float* __restrict__ bconv, const float* __restrict__ x,
              __bf16* __restrict__ rbf, int NN)
{
    int d = blockIdx.x * (blockDim.x >> 5) + (threadIdx.x >> 5);
    if (d >= NN) return;
    int lane = threadIdx.x & 31;
    float dd = dis[d];

    // self-loop: dis[d]^2 * h[d]
    float4 acc = ((const float4*)(h + (size_t)d * 128))[lane];
    acc.x *= dd * dd; acc.y *= dd * dd; acc.z *= dd * dd; acc.w *= dd * dd;

    int beg = rowStart[d], end = rowStart[d + 1];
    int e = beg;
    for (; e + 1 < end; e += 2) {
        int s0 = edgeSrc[e];
        int s1 = edgeSrc[e + 1];
        float nm0 = dis[s0] * dd;
        float nm1 = dis[s1] * dd;
        float4 v0 = ((const float4*)(h + (size_t)s0 * 128))[lane];
        float4 v1 = ((const float4*)(h + (size_t)s1 * 128))[lane];
        acc.x += nm0 * v0.x + nm1 * v1.x;
        acc.y += nm0 * v0.y + nm1 * v1.y;
        acc.z += nm0 * v0.z + nm1 * v1.z;
        acc.w += nm0 * v0.w + nm1 * v1.w;
    }
    if (e < end) {
        int s = edgeSrc[e];
        float nm = dis[s] * dd;
        float4 v = ((const float4*)(h + (size_t)s * 128))[lane];
        acc.x += nm * v.x; acc.y += nm * v.y;
        acc.z += nm * v.z; acc.w += nm * v.w;
    }

    float4 bb = ((const float4*)bconv)[lane];
    float4 xv = ((const float4*)(x + (size_t)d * 128))[lane];
    __bf16* o = rbf + (size_t)d * 128 + lane * 4;
    o[0] = (__bf16)(fmaxf(acc.x + bb.x, 0.0f) + xv.x);
    o[1] = (__bf16)(fmaxf(acc.y + bb.y, 0.0f) + xv.y);
    o[2] = (__bf16)(fmaxf(acc.z + bb.z, 0.0f) + xv.z);
    o[3] = (__bf16)(fmaxf(acc.w + bb.w, 0.0f) + xv.w);
}

// out[node] = dot(h2[node,0:256], W3) + b3 — one wave per node
__global__ __launch_bounds__(256)
void k_final(const __bf16* __restrict__ h2, const float* __restrict__ W3,
             const float* __restrict__ b3, float* out, int M)
{
    int wid = blockIdx.x * (blockDim.x >> 5) + (threadIdx.x >> 5);
    if (wid >= M) return;
    int lane = threadIdx.x & 31;
    const __bf16* row = h2 + (size_t)wid * 256;
    float s = 0.0f;
#pragma unroll
    for (int j = 0; j < 8; ++j) {
        int k = lane + j * 32;
        s += (float)row[k] * W3[k];
    }
    for (int off = 16; off; off >>= 1) s += __shfl_down(s, off, 32);
    if (lane == 0) out[wid] = s + b3[0];
}

// ---------------- launch ----------------

extern "C" void kernel_launch(void* const* d_in, const int* in_sizes, int n_in,
                              void* d_out, int out_size, void* d_ws, size_t ws_size,
                              hipStream_t stream) {
    const float*     x  = (const float*)d_in[0];
    const long long* ei = (const long long*)d_in[1];
    const float*     Wc = (const float*)d_in[2];
    const float*     bc = (const float*)d_in[3];
    const float*     W1 = (const float*)d_in[4];
    const float*     b1 = (const float*)d_in[5];
    const float*     W2 = (const float*)d_in[6];
    const float*     b2 = (const float*)d_in[7];
    const float*     W3 = (const float*)d_in[8];
    const float*     b3 = (const float*)d_in[9];
    float* out = (float*)d_out;

    const int NN = in_sizes[0] / 128;   // 50000 (== 3125 * 16)
    const int E  = in_sizes[1] / 2;     // 800000
    const size_t NX = (size_t)NN * 128;
    const int nb = (NN + 255) / 256;    // scan blocks (196 <= 256)

    char* ws = (char*)d_ws;
    size_t off = 0;
    auto alloc = [&](size_t bytes) { size_t o = off; off = (off + bytes + 255) & ~(size_t)255; return o; };
    int*    cnt      = (int*)(ws + alloc((size_t)NN * 4));
    int*    rowStart = (int*)(ws + alloc((size_t)(NN + 1) * 4));
    int*    cursor   = (int*)(ws + alloc((size_t)NN * 4));
    int*    bsums    = (int*)(ws + alloc(256 * 4));
    int*    edgeSrc  = (int*)(ws + alloc((size_t)E * 4));
    float*  dis      = (float*)(ws + alloc((size_t)NN * 4));
    __bf16* wpc      = (__bf16*)(ws + alloc(128 * 128 * 2));
    __bf16* wp1      = (__bf16*)(ws + alloc(128 * 256 * 2));
    __bf16* wp2      = (__bf16*)(ws + alloc(256 * 256 * 2));
    __bf16* xbf      = (__bf16*)(ws + alloc(NX * 2));
    float*  h        = (float*)(ws + alloc(NX * 4));
    __bf16* h2       = (__bf16*)(ws + alloc((size_t)NN * 256 * 2));
    __bf16* rbf = xbf;            // x-bf16 dead after GEMM1
    __bf16* h1  = (__bf16*)h;     // h dead after gather

    const int B = 256;

    // CSR build + normalization
    k_zero_i32 <<<(NN + 255) / 256, B, 0, stream>>>(cnt, NN);
    k_count    <<<(E + 255) / 256, B, 0, stream>>>(ei, cnt, E);
    k_dis      <<<(NN + 255) / 256, B, 0, stream>>>(cnt, dis, NN);
    k_scan_block<<<nb, B, 0, stream>>>(cnt, rowStart, bsums, NN);
    k_scan_tops <<<1, B, 0, stream>>>(bsums, nb);
    k_scan_add  <<<(NN + 256) / 256, B, 0, stream>>>(rowStart, bsums, NN, E);
    k_copy_i32  <<<(NN + 255) / 256, B, 0, stream>>>(rowStart, cursor, NN);
    k_place     <<<(E + 255) / 256, B, 0, stream>>>(ei, cursor, edgeSrc, E);

    // bf16 conversions + weight repack
    k_cvt_bf16<<<(int)((NX + 255) / 256), B, 0, stream>>>(x, xbf, NX);
    k_pack_w<<<(128 * 128 + 255) / 256, B, 0, stream>>>(Wc, wpc, 128, 128);
    k_pack_w<<<(128 * 256 + 255) / 256, B, 0, stream>>>(W1, wp1, 128, 256);
    k_pack_w<<<(256 * 256 + 255) / 256, B, 0, stream>>>(W2, wp2, 256, 256);

    // GEMM1: h = x @ W_conv   (fp32 out, bias applied in gather)
    {
        int waves = (NN / 16) * (128 / 64);
        k_gemm<<<(waves + 7) / 8, B, 0, stream>>>(xbf, wpc, nullptr, h, nullptr,
                                                  NN, 128, 128, 0);
    }

    // fused aggregation + bias + relu + residual -> bf16
    k_gather<<<(NN + 7) / 8, B, 0, stream>>>(rowStart, edgeSrc, dis, h, bc, x, rbf, NN);

    // GEMM2: h1 = relu(r @ W1 + b1)   [NN,128]x[128,256] -> bf16
    {
        int waves = (NN / 16) * (256 / 64);
        k_gemm<<<(waves + 7) / 8, B, 0, stream>>>(rbf, wp1, b1, nullptr, h1,
                                                  NN, 256, 128, 1);
    }
    // GEMM3: h2 = relu(h1 @ W2 + b2)  [NN,256]x[256,256] -> bf16
    {
        int waves = (NN / 16) * (256 / 64);
        k_gemm<<<(waves + 7) / 8, B, 0, stream>>>(h1, wp2, b2, nullptr, h2,
                                                  NN, 256, 256, 1);
    }
    // final projection to scalar
    k_final<<<(NN + 7) / 8, B, 0, stream>>>(h2, W3, b3, out, NN);
}
